// RPN_87505663689217
// MI455X (gfx1250) — matrix operand order repaired and checked by reading, not compiled, not run on hardware
//
#include <hip/hip_runtime.h>
#include <hip/hip_bf16.h>

// ---------------------------------------------------------------------------
// CDNA5 (gfx1250) RPN implementation.
//   k_feat_nhwc : f32 NCHW -> bf16 NHWC activation repack
//   k_pack_w    : f32 OIHW conv weights -> bf16 WMMA B-fragment layout
//   k_conv_wmma : implicit-GEMM 3x3 conv, 64x64 macro-tile per wave,
//                 16x v_wmma_f32_16x16x32_bf16 per K-chunk (register reuse)
//   k_heads     : 1x1 cls/bbox heads + paired softmax + anchor decode/clip
//   k_sort      : per-batch 16K-key bitonic sort fully inside 128KB LDS
//   k_nms       : greedy NMS (== reference argmax/suppress scan) -> rois
// ---------------------------------------------------------------------------

typedef __attribute__((ext_vector_type(16))) __bf16        v16bf;
typedef __attribute__((ext_vector_type(8)))  float         v8f;
typedef __attribute__((ext_vector_type(4)))  unsigned int  u32x4;

union Frag { v16bf v; u32x4 q[2]; };

__device__ __forceinline__ unsigned short f2bf(float f) {
  unsigned int u = __float_as_uint(f);
  u += 0x7FFFu + ((u >> 16) & 1u);          // round-to-nearest-even
  return (unsigned short)(u >> 16);
}

// ------------------------- activations: NCHW f32 -> NHWC bf16 --------------
__global__ void k_feat_nhwc(const float* __restrict__ in,
                            unsigned short* __restrict__ out) {
  int i = blockIdx.x * blockDim.x + threadIdx.x;      // over 2*4096*1024
  if (i >= 2 * 4096 * 1024) return;
  int c = i & 1023;
  int m = (i >> 10) & 4095;
  int b = i >> 22;
  float v = in[((b << 10) + c) * 4096 + m];
  out[i] = f2bf(v);
}

// ------------------------- weights -> WMMA B-fragment order ----------------
// wfrag[nt][kpos][cc][lane][e] : B 32x16 bf16 fragment layout:
//   N = lane%16 (output channel), K = e<8 ? (lane/16)*8 + e : 16 + (lane/16)*8 + e-8
__global__ void k_pack_w(const float* __restrict__ Wc,
                         unsigned short* __restrict__ wf) {
  int f = blockIdx.x * blockDim.x + threadIdx.x;      // 32*9*32*32*16
  if (f >= 32 * 9 * 32 * 32 * 16) return;
  int e    = f & 15;  int r = f >> 4;
  int lane = r & 31;  r >>= 5;
  int cc   = r & 31;  r >>= 5;
  int kpos = r % 9;
  int nt   = r / 9;
  int half = lane >> 4, l16 = lane & 15;
  int co = nt * 16 + l16;
  int c  = cc * 32 + (e < 8 ? half * 8 + e : 16 + half * 8 + (e - 8));
  int ky = kpos / 3, kx = kpos % 3;
  wf[f] = f2bf(Wc[((co * 1024 + c) * 3 + ky) * 3 + kx]);
}

// ------------------------- 3x3 conv as implicit GEMM (WMMA) ----------------
// One wave -> 64 pixels (one full image row) x 64 cout macro-tile:
// 16 accumulators, 4 A-frags + 4 B-frags per 32-deep K chunk -> 16 WMMAs.
// Per-tile global traffic 0.5KB/K-chunk (4x better than 1 tile/wave).
__global__ void __launch_bounds__(256) k_conv_wmma(
    const unsigned short* __restrict__ feat,
    const unsigned short* __restrict__ wfrag,
    const float* __restrict__ bconv,
    float* __restrict__ xout) {
  const int lane = threadIdx.x & 31;
  const int wave = (blockIdx.x * blockDim.x + threadIdx.x) >> 5;  // < 1024
  const int nt64 = wave & 7;           // 64-wide cout tile
  const int h    = (wave >> 3) & 63;   // image row (64 pixels == M tile)
  const int b    = wave >> 9;
  const int half = lane >> 4;
  const int l16  = lane & 15;

  v8f acc[4][4];                        // [m-subtile][n-subtile]
  #pragma unroll
  for (int ms = 0; ms < 4; ++ms)
    #pragma unroll
    for (int ns = 0; ns < 4; ++ns) acc[ms][ns] = v8f{};

  for (int kpos = 0; kpos < 9; ++kpos) {
    const int dy = kpos / 3 - 1;
    const int dx = kpos % 3 - 1;
    const int y = h + dy;
    if ((unsigned)y >= 64u) continue;             // wave-uniform skip
    const unsigned short* prow =
        feat + ((((b << 6) + y) << 6) << 10) + half * 8;   // + x*1024 later
    for (int cc = 0; cc < 32; ++cc) {
      Frag bf[4];
      #pragma unroll
      for (int ns = 0; ns < 4; ++ns) {
        const unsigned short* pB =
            wfrag + (((nt64 << 2) + ns) * 9 + kpos) * 16384 + lane * 16 +
            cc * 512;
        bf[ns].q[0] = *reinterpret_cast<const u32x4*>(pB);
        bf[ns].q[1] = *reinterpret_cast<const u32x4*>(pB + 8);
      }
      #pragma unroll
      for (int ms = 0; ms < 4; ++ms) {
        const int x = (ms << 4) + l16 + dx;       // per-lane halo predicate
        Frag a;
        if ((unsigned)x < 64u) {
          const unsigned short* pA = prow + (x << 10) + (cc << 5);
          a.q[0] = *reinterpret_cast<const u32x4*>(pA);       // K=half*8..
          a.q[1] = *reinterpret_cast<const u32x4*>(pA + 16);  // K=16+half*8..
        } else {
          a.q[0] = u32x4{0u, 0u, 0u, 0u};
          a.q[1] = u32x4{0u, 0u, 0u, 0u};
        }
        #pragma unroll
        for (int ns = 0; ns < 4; ++ns)
          acc[ms][ns] = __builtin_amdgcn_wmma_f32_16x16x32_bf16(
              false, a.v, false, bf[ns].v, (short)0, acc[ms][ns], false,
              false);
      }
    }
  }
  // epilogue: bias + ReLU, C/D layout M = i + 8*half, N = l16
  #pragma unroll
  for (int ms = 0; ms < 4; ++ms) {
    const int pbase = (h << 6) + (ms << 4);       // first pixel of subtile
    #pragma unroll
    for (int ns = 0; ns < 4; ++ns) {
      const int co = (nt64 << 6) + (ns << 4) + l16;
      const float bias = bconv[co];
      float* po = xout + ((b << 12) + pbase) * 512 + co;
      #pragma unroll
      for (int i = 0; i < 8; ++i) {
        float v = acc[ms][ns][i] + bias;
        po[(i + half * 8) * 512] = v > 0.f ? v : 0.f;
      }
    }
  }
}

// ------------------------- heads + softmax + decode ------------------------
__global__ void __launch_bounds__(256) k_heads(
    const float* __restrict__ xout,
    const float* __restrict__ Wcls, const float* __restrict__ bcls,
    const float* __restrict__ Wbb,  const float* __restrict__ bbb,
    const float* __restrict__ meta,
    float* __restrict__ scores, float* __restrict__ props) {
  __shared__ float wsm[20 * 512];
  __shared__ float bsm[20];
  for (int i = threadIdx.x; i < 20 * 512; i += blockDim.x) {
    int o = i >> 9, c = i & 511;
    wsm[i] = (o < 4) ? Wcls[o * 512 + c] : Wbb[(o - 4) * 512 + c];
  }
  if (threadIdx.x < 20)
    bsm[threadIdx.x] = (threadIdx.x < 4) ? bcls[threadIdx.x]
                                         : bbb[threadIdx.x - 4];
  __syncthreads();

  int t = blockIdx.x * blockDim.x + threadIdx.x;    // one thread per pixel
  if (t >= 2 * 4096) return;
  int b = t >> 12, m = t & 4095;
  const float* xr = xout + (size_t)t * 512;
  float s[20];
  #pragma unroll
  for (int o = 0; o < 20; ++o) s[o] = bsm[o];
  for (int c = 0; c < 512; ++c) {
    float xv = xr[c];
    #pragma unroll
    for (int o = 0; o < 20; ++o) s[o] += xv * wsm[o * 512 + c];
  }
  // reshape(B,2,A*H/2,W) softmax over axis 1  == pairwise softmax (a, a+2)
  float p[4];
  #pragma unroll
  for (int a2 = 0; a2 < 2; ++a2) {
    float s0 = s[a2], s1 = s[a2 + 2];
    float mx = fmaxf(s0, s1);
    float e0 = __expf(s0 - mx), e1 = __expf(s1 - mx);
    float inv = 1.f / (e0 + e1);
    p[a2] = e0 * inv;
    p[a2 + 2] = e1 * inv;
  }
  const float adim[4] = {64.f, 128.f, 256.f, 512.f};
  int hh = m >> 6, ww = m & 63;
  float cx = ww * 16.f + 8.f, cy = hh * 16.f + 8.f;  // anchor centers
  float im_h = meta[b * 3 + 0], im_w = meta[b * 3 + 1], scl = meta[b * 3 + 2];
  float min_sz = 16.f * scl;
  #pragma unroll
  for (int a = 0; a < 4; ++a) {
    float wa = adim[a];                                  // square anchors
    float d0 = s[4 + a * 4 + 0], d1 = s[4 + a * 4 + 1];
    float d2 = s[4 + a * 4 + 2], d3 = s[4 + a * 4 + 3];
    float pcx = d0 * wa + cx, pcy = d1 * wa + cy;
    float pw = __expf(d2) * wa, ph = __expf(d3) * wa;
    float x1 = pcx - 0.5f * pw, y1 = pcy - 0.5f * ph;
    float x2 = pcx + 0.5f * pw, y2 = pcy + 0.5f * ph;
    x1 = fminf(fmaxf(x1, 0.f), im_w - 1.f);
    y1 = fminf(fmaxf(y1, 0.f), im_h - 1.f);
    x2 = fminf(fmaxf(x2, 0.f), im_w - 1.f);
    y2 = fminf(fmaxf(y2, 0.f), im_h - 1.f);
    bool keep = (x2 - x1 + 1.f >= min_sz) && (y2 - y1 + 1.f >= min_sz);
    int idx = (m << 2) + a;
    scores[b * 16384 + idx] = keep ? p[a] : -__builtin_inff();
    float* pp = props + (((size_t)b * 16384 + idx) << 2);
    pp[0] = x1; pp[1] = y1; pp[2] = x2; pp[3] = y2;
  }
}

// ------------------------- in-LDS bitonic top-k sort -----------------------
// 16384 keys * 8B = 128KB fits in one WGP's 320KB LDS (CDNA5-only capacity).
// Key = sortable(score)<<32 | (~idx)  -> descending sort == stable top_k.
__global__ void __launch_bounds__(1024) k_sort(const float* __restrict__ scores,
                                               int* __restrict__ sidx) {
  extern __shared__ unsigned long long keys[];
  const int b = blockIdx.x, tid = threadIdx.x;
  for (int i = tid; i < 16384; i += 1024) {
    unsigned int u = __float_as_uint(scores[b * 16384 + i]);
    u = (u & 0x80000000u) ? ~u : (u | 0x80000000u);
    keys[i] = ((unsigned long long)u << 32) |
              (unsigned long long)(0xFFFFFFFFu - (unsigned)i);
  }
  __syncthreads();
  for (int k = 2; k <= 16384; k <<= 1) {
    for (int j = k >> 1; j > 0; j >>= 1) {
      for (int i = tid; i < 16384; i += 1024) {
        int ixj = i ^ j;
        if (ixj > i) {
          unsigned long long a = keys[i], c = keys[ixj];
          bool sw = ((i & k) == 0) ? (a < c) : (a > c);   // descending
          if (sw) { keys[i] = c; keys[ixj] = a; }
        }
      }
      __syncthreads();
    }
  }
  for (int i = tid; i < 6000; i += 1024)
    sidx[b * 6000 + i] =
        (int)(0xFFFFFFFFu - (unsigned)(keys[i] & 0xFFFFFFFFull));
}

// ------------------------- greedy NMS (== reference scan) ------------------
__global__ void __launch_bounds__(1024) k_nms(const float* __restrict__ scores,
                                              const float* __restrict__ props,
                                              const int* __restrict__ sidx,
                                              float* __restrict__ out) {
  extern __shared__ float lds[];
  float* bx1 = lds;
  float* by1 = lds + 6000;
  float* bx2 = lds + 12000;
  float* by2 = lds + 18000;
  float* bar = lds + 24000;
  float* bsc = lds + 30000;
  unsigned char* flg = (unsigned char*)(lds + 36000);
  __shared__ int s_j, s_cur;
  const int b = blockIdx.x, tid = threadIdx.x;
  for (int i = tid; i < 6000; i += 1024) {
    int idx = sidx[b * 6000 + i];
    const float* pp = props + (((size_t)b * 16384 + idx) << 2);
    float x1 = pp[0], y1 = pp[1], x2 = pp[2], y2 = pp[3];
    bx1[i] = x1; by1[i] = y1; bx2[i] = x2; by2[i] = y2;
    bar[i] = (x2 - x1 + 1.f) * (y2 - y1 + 1.f);
    bsc[i] = scores[b * 16384 + idx];
    flg[i] = 0;
  }
  if (tid == 0) s_cur = 0;
  __syncthreads();
  for (int t = 0; t < 300; ++t) {
    if (tid == 0) {
      int i = s_cur;
      while (i < 6000 && flg[i]) ++i;
      s_cur = i;
      s_j = (i < 6000 && bsc[i] != -__builtin_inff()) ? i : -1;
    }
    __syncthreads();
    int j = s_j;
    int ref = (j >= 0) ? j : 0;          // all -inf => argmax returns pos 0
    if (tid == 0) {
      float* ro = out + ((size_t)b * 300 + t) * 5;
      ro[0] = (float)b;
      ro[1] = bx1[ref]; ro[2] = by1[ref]; ro[3] = bx2[ref]; ro[4] = by2[ref];
    }
    if (j >= 0) {
      float rx1 = bx1[j], ry1 = by1[j], rx2 = bx2[j], ry2 = by2[j];
      float ra = bar[j];
      for (int i = tid; i < 6000; i += 1024) {
        if (!flg[i]) {
          float xx1 = fmaxf(rx1, bx1[i]);
          float yy1 = fmaxf(ry1, by1[i]);
          float xx2 = fminf(rx2, bx2[i]);
          float yy2 = fminf(ry2, by2[i]);
          float inter = fmaxf(xx2 - xx1 + 1.f, 0.f) *
                        fmaxf(yy2 - yy1 + 1.f, 0.f);
          float iou = inter / (ra + bar[i] - inter);
          if (iou > 0.7f) flg[i] = 1;
        }
      }
    }
    __syncthreads();
  }
}

// ---------------------------------------------------------------------------
extern "C" void kernel_launch(void* const* d_in, const int* in_sizes, int n_in,
                              void* d_out, int out_size, void* d_ws,
                              size_t ws_size, hipStream_t stream) {
  const float* base  = (const float*)d_in[0];
  const float* meta  = (const float*)d_in[1];
  // d_in[2] = gt_boxes (unused by forward pass)
  const float* Wconv = (const float*)d_in[3];
  const float* bconv = (const float*)d_in[4];
  const float* Wcls  = (const float*)d_in[5];
  const float* bcls  = (const float*)d_in[6];
  const float* Wbb   = (const float*)d_in[7];
  const float* bbb   = (const float*)d_in[8];
  float* out = (float*)d_out;

  char* ws = (char*)d_ws;
  unsigned short* feat  = (unsigned short*)(ws);                 // 16.0 MiB
  unsigned short* wfrag = (unsigned short*)(ws + 16777216);      //  9.0 MiB
  float* xout   = (float*)(ws + 16777216 + 9437184);             // 16.0 MiB
  float* scores = (float*)(ws + 16777216 + 9437184 + 16777216);  // 128 KiB
  float* props  = scores + 2 * 16384;                            // 512 KiB
  int*   sidx   = (int*)(props + 2 * 16384 * 4);                 //  48 KiB

  k_feat_nhwc<<<(2 * 4096 * 1024 + 255) / 256, 256, 0, stream>>>(base, feat);
  k_pack_w<<<(32 * 9 * 32 * 32 * 16 + 255) / 256, 256, 0, stream>>>(Wconv,
                                                                    wfrag);
  // 1024 waves (2 batches x 64 rows x 8 cout-tiles), 8 waves per block
  k_conv_wmma<<<128, 256, 0, stream>>>(feat, wfrag, bconv, xout);
  k_heads<<<32, 256, 0, stream>>>(xout, Wcls, bcls, Wbb, bbb, meta, scores,
                                  props);
  k_sort<<<2, 1024, 16384 * sizeof(unsigned long long), stream>>>(scores,
                                                                  sidx);
  k_nms<<<2, 1024, 36000 * 4 + 6016, stream>>>(scores, props, sidx, out);
}
